// PrefixKVGenerator_23648089932346
// MI455X (gfx1250) — matrix operand order repaired and checked by reading, not compiled, not run on hardware
//
#include <hip/hip_runtime.h>
#include <hip/hip_bf16.h>

// ---------------- problem constants ----------------
constexpr int D_SUMMARY = 128;
constexpr int N_LAYERS  = 24;
constexpr int N_HEADS   = 16;
constexpr int D_HEAD    = 64;
constexpr int HIDDEN    = 512;
constexpr int ACTIVE_MIN= 12;
constexpr int ROT       = 16;
constexpr int Bsz       = 16;
constexpr int Psz       = 128;
constexpr int L_ACT     = N_LAYERS - ACTIVE_MIN;     // 12
constexpr int OUT_DIM   = N_HEADS * 2 * D_HEAD;      // 2048
constexpr int Mrows     = Bsz * Psz;                 // 2048
constexpr int LDS_LD    = 40;                        // 32 halves data + 8 pad (80B rows)

typedef __attribute__((ext_vector_type(16))) _Float16 v16h;
typedef __attribute__((ext_vector_type(8)))  float    v8f;

#if defined(__has_builtin)
#if __has_builtin(__builtin_amdgcn_global_load_async_to_lds_b128) && \
    __has_builtin(__builtin_amdgcn_s_wait_asynccnt)
#define USE_ASYNC_LDS 1
#endif
#endif

// generic-vector int4 matching clang's V4i builtin pointee
typedef int vs4i __attribute__((vector_size(16)));
typedef __attribute__((address_space(1))) vs4i* gvec_p;
typedef __attribute__((address_space(3))) vs4i* lvec_p;

// 16-byte staged copy global -> LDS
__device__ __forceinline__ void copy16(const _Float16* __restrict__ g,
                                       _Float16* __restrict__ l) {
#if defined(USE_ASYNC_LDS)
  __builtin_amdgcn_global_load_async_to_lds_b128(
      (gvec_p)(void*)const_cast<_Float16*>(g),
      (lvec_p)(void*)l, 0, 0);
#else
  int4 v = *(const int4*)g;
  *(int4*)l = v;
#endif
}

template <int N>
__device__ __forceinline__ void wait_stage() {
#if defined(USE_ASYNC_LDS)
  __builtin_amdgcn_s_wait_asynccnt(N);
#endif
}

// ---------------- WMMA fragment loaders (contiguous-k layouts, same for A and B, wave32) --------
// A 16x32 f16 row-major: lane M = lane&15 ; K = e + 16*(lane>=16)
__device__ __forceinline__ v16h load_A_frag(const _Float16* __restrict__ src, int ld,
                                            int row0, int k0, int lane) {
  const int m  = lane & 15;
  const int kb = k0 + ((lane >> 4) << 4);
  const _Float16* p = src + (size_t)(row0 + m) * ld + kb;
  v16h a;
#pragma unroll
  for (int e = 0; e < 8; ++e) a[e] = p[e];
#pragma unroll
  for (int e = 0; e < 8; ++e) a[8 + e] = p[8 + e];
  return a;
}

// B 32x16 (KxN) from N-major rows: lane N = lane&15 ; K = e + 16*(lane>=16)
__device__ __forceinline__ v16h load_B_frag(const _Float16* __restrict__ src, int ldk,
                                            int k0, int n0, int lane) {
  const int n  = lane & 15;
  const int kb = k0 + ((lane >> 4) << 4);
  const _Float16* p = src + (size_t)(n0 + n) * ldk + kb;
  v16h b;
#pragma unroll
  for (int e = 0; e < 16; ++e) b[e] = p[e];
  return b;
}

// same layouts but from an LDS tile with row stride LDS_LD
__device__ __forceinline__ v16h lds_A_frag(const _Float16* s, int row0, int lane) {
  const _Float16* p = s + (row0 + (lane & 15)) * LDS_LD + ((lane >> 4) << 4);
  v16h a;
#pragma unroll
  for (int e = 0; e < 8; ++e) a[e] = p[e];
#pragma unroll
  for (int e = 0; e < 8; ++e) a[8 + e] = p[8 + e];
  return a;
}
__device__ __forceinline__ v16h lds_B_frag(const _Float16* s, int row0, int lane) {
  const _Float16* p = s + (row0 + (lane & 15)) * LDS_LD + ((lane >> 4) << 4);
  v16h b;
#pragma unroll
  for (int e = 0; e < 16; ++e) b[e] = p[e];
  return b;
}

__device__ __forceinline__ v8f wmma_step(v16h a, v16h b, v8f c) {
  return __builtin_amdgcn_wmma_f32_16x16x32_f16(false, a, false, b,
                                                (short)0, c, false, false);
}

// ---------------- prep 1: f32 -> f16 straight convert (for S) --------------
__global__ void cvt_f16_kernel(const float* __restrict__ src,
                               _Float16* __restrict__ dst, int n) {
  int i = (blockIdx.x * blockDim.x + threadIdx.x) * 4;
  if (i < n) {
    float4 v = *(const float4*)(src + i);
    dst[i + 0] = (_Float16)v.x;
    dst[i + 1] = (_Float16)v.y;
    dst[i + 2] = (_Float16)v.z;
    dst[i + 3] = (_Float16)v.w;
  }
}

// ---------------- prep 2: per-layer convert + transpose (K,N)->(N,K) -------
template <int KDIM, int NDIM>
__global__ void cvt_transpose_kernel(const float* __restrict__ src,
                                     _Float16* __restrict__ dst) {
  __shared__ _Float16 tile[32][33];
  const int l  = blockIdx.z;
  const float* s = src + (size_t)l * KDIM * NDIM;
  _Float16*    d = dst + (size_t)l * KDIM * NDIM;
  const int tx = threadIdx.x, ty = threadIdx.y;
  const int k0 = blockIdx.y * 32, n0 = blockIdx.x * 32;
#pragma unroll
  for (int i = ty; i < 32; i += 8)
    tile[i][tx] = (_Float16)s[(size_t)(k0 + i) * NDIM + (n0 + tx)];
  __syncthreads();
#pragma unroll
  for (int i = ty; i < 32; i += 8)
    d[(size_t)(n0 + i) * KDIM + (k0 + tx)] = tile[tx][i];
}

// ---------------- kernel: zero the first ACTIVE_MIN layers ------------------
__global__ void zero_prefix_kernel(float4* __restrict__ out, int n4) {
  int i = blockIdx.x * blockDim.x + threadIdx.x;
  if (i < n4) out[i] = make_float4(0.f, 0.f, 0.f, 0.f);
}

// ---------------- GEMM1: H = relu(Sh @ W1t^T + b1), f16 out -----------------
__global__ void gemm1_relu_kernel(const _Float16* __restrict__ Sh,
                                  const _Float16* __restrict__ W1t, // (L,512,128)
                                  const float* __restrict__ b1,
                                  _Float16* __restrict__ H) {
  const int wave = (blockIdx.x * blockDim.x + threadIdx.x) >> 5;
  const int lane = threadIdx.x & 31;
  const int l  = wave >> 10;
  const int t  = wave & 1023;
  const int m0 = (t >> 4) * 32;
  const int n0 = (t & 15) * 32;

  const _Float16* Wl = W1t + (size_t)l * HIDDEN * D_SUMMARY;
  v8f acc[2][2] = {};
#pragma unroll
  for (int k0 = 0; k0 < D_SUMMARY; k0 += 32) {
    v16h a[2], b[2];
#pragma unroll
    for (int i = 0; i < 2; ++i) a[i] = load_A_frag(Sh, D_SUMMARY, m0 + 16 * i, k0, lane);
#pragma unroll
    for (int j = 0; j < 2; ++j) b[j] = load_B_frag(Wl, D_SUMMARY, k0, n0 + 16 * j, lane);
#pragma unroll
    for (int i = 0; i < 2; ++i)
#pragma unroll
      for (int j = 0; j < 2; ++j) acc[i][j] = wmma_step(a[i], b[j], acc[i][j]);
  }

  _Float16* outH = H + (size_t)l * Mrows * HIDDEN;
#pragma unroll
  for (int j = 0; j < 2; ++j) {
    const int n = n0 + 16 * j + (lane & 15);
    const float bias = b1[l * HIDDEN + n];
#pragma unroll
    for (int i = 0; i < 2; ++i)
#pragma unroll
      for (int r = 0; r < 8; ++r) {
        const int m = m0 + 16 * i + r + ((lane >> 4) << 3);
        float v = acc[i][j][r] + bias;
        v = v > 0.f ? v : 0.f;
        outH[(size_t)m * HIDDEN + n] = (_Float16)v;
      }
  }
}

// ---------------- GEMM2: 128x128 block tile, LDS double-buffer --------------
// 8 waves = 4(M) x 2(N), each wave 32x64 (2x4 WMMA tiles)
__global__ void gemm2_kv_kernel(const _Float16* __restrict__ H,
                                const _Float16* __restrict__ W2t, // (L,2048,512)
                                const float* __restrict__ b2,
                                const int* __restrict__ positions,
                                float* __restrict__ out) {
  __shared__ _Float16 sA[2][128 * LDS_LD];   // 2 x 10 KB
  __shared__ _Float16 sB[2][128 * LDS_LD];   // 2 x 10 KB

  const int tid  = threadIdx.x;
  const int lane = tid & 31;
  const int wid  = tid >> 5;
  const int wm   = wid >> 1;                 // 0..3
  const int wn   = wid & 1;                  // 0..1

  const int bx = blockIdx.x;
  const int l  = bx >> 8;                    // 256 blocks/layer
  const int t  = bx & 255;
  const int blkM = (t >> 4) * 128;
  const int blkN = (t & 15) * 128;

  const _Float16* gA = H   + (size_t)l * Mrows * HIDDEN   + (size_t)blkM * HIDDEN;
  const _Float16* gB = W2t + (size_t)l * OUT_DIM * HIDDEN + (size_t)blkN * HIDDEN;

  // stage one 128x32 k-chunk of A and B (4 x b128 per thread)
  auto stage = [&](int buf, int k0) {
#pragma unroll
    for (int h = 0; h < 2; ++h) {
      const int c   = tid + 256 * h;          // 0..511
      const int row = c >> 2;
      const int kp  = (c & 3) * 8;            // halves
      copy16(gA + (size_t)row * HIDDEN + k0 + kp, &sA[buf][row * LDS_LD + kp]);
      copy16(gB + (size_t)row * HIDDEN + k0 + kp, &sB[buf][row * LDS_LD + kp]);
    }
  };

  v8f acc[2][4] = {};
  stage(0, 0);
  constexpr int NCHUNK = HIDDEN / 32;         // 16
  for (int it = 0; it < NCHUNK; ++it) {
    const int cur = it & 1;
    if (it + 1 < NCHUNK) {
      stage(cur ^ 1, (it + 1) * 32);
      wait_stage<4>();                        // own chunk-it issues complete
    } else {
      wait_stage<0>();
    }
    __syncthreads();                          // all waves' chunk-it data in LDS

    v16h a[2], b[4];
#pragma unroll
    for (int i = 0; i < 2; ++i) a[i] = lds_A_frag(sA[cur], wm * 32 + 16 * i, lane);
#pragma unroll
    for (int j = 0; j < 4; ++j) b[j] = lds_B_frag(sB[cur], wn * 64 + 16 * j, lane);
#pragma unroll
    for (int i = 0; i < 2; ++i)
#pragma unroll
      for (int j = 0; j < 4; ++j) acc[i][j] = wmma_step(a[i], b[j], acc[i][j]);

    __syncthreads();                          // done reading before overwrite
  }

  // ---- epilogue: bias + RoPE + transposed scatter
  const int m0 = blkM + wm * 32;
  const int n0base = blkN + wn * 64;
#pragma unroll
  for (int j = 0; j < 4; ++j) {
    const int ntile = n0base + 16 * j;
    const int n     = ntile + (lane & 15);
    const int head  = n >> 7;
    const int kvsel = (n >> 6) & 1;                    // tile-uniform
    const int d     = n & 63;
    const bool do_rope = (kvsel == 0) && ((ntile & 63) == 0);
    const float bias = b2[l * OUT_DIM + n];
    const float inv_freq = __powf(10000.0f, -(float)(d & 7) * 0.125f);

#pragma unroll
    for (int i = 0; i < 2; ++i)
#pragma unroll
      for (int r = 0; r < 8; ++r) {
        const int m  = m0 + 16 * i + r + ((lane >> 4) << 3);
        const int bb = m >> 7;
        const int p  = m & 127;
        float v = acc[i][j][r] + bias;

        const float other = __shfl_xor(v, 8, 32);      // RoPE pair = lane^8
        float val = v;
        if (do_rope) {
          const float ang = (float)positions[p] * inv_freq;
          float s, c;
          __sincosf(ang, &s, &c);
          val = (d < 8) ? (v * c - other * s) : (v * c + other * s);
        }

        const size_t idx =
            (((((size_t)(ACTIVE_MIN + l) * 2 + kvsel) * Bsz + bb) * N_HEADS + head)
                 * Psz + p) * D_HEAD + d;
        out[idx] = val;
      }
  }
}

// ---------------- launcher ---------------------------------------------------
extern "C" void kernel_launch(void* const* d_in, const int* in_sizes, int n_in,
                              void* d_out, int out_size, void* d_ws, size_t ws_size,
                              hipStream_t stream) {
  const float* S         = (const float*)d_in[0];
  const int*   positions = (const int*)  d_in[1];
  const float* W1        = (const float*)d_in[2];
  const float* b1        = (const float*)d_in[3];
  const float* W2        = (const float*)d_in[4];
  const float* b2        = (const float*)d_in[5];
  float*       out       = (float*)d_out;

  // workspace carve-up (f16): Sh | W1t | W2t | H  (~52.4 MB)
  _Float16* Sh  = (_Float16*)d_ws;
  _Float16* W1t = Sh  + (size_t)Mrows * D_SUMMARY;
  _Float16* W2t = W1t + (size_t)L_ACT * HIDDEN * D_SUMMARY;
  _Float16* H   = W2t + (size_t)L_ACT * OUT_DIM * HIDDEN;

  {
    const int n = Mrows * D_SUMMARY;
    cvt_f16_kernel<<<n / 4 / 256, 256, 0, stream>>>(S, Sh, n);
  }
  {
    dim3 blk(32, 8);
    dim3 g1(HIDDEN / 32, D_SUMMARY / 32, L_ACT);
    cvt_transpose_kernel<D_SUMMARY, HIDDEN><<<g1, blk, 0, stream>>>(W1, W1t);
    dim3 g2(OUT_DIM / 32, HIDDEN / 32, L_ACT);
    cvt_transpose_kernel<HIDDEN, OUT_DIM><<<g2, blk, 0, stream>>>(W2, W2t);
  }
  {
    const int n4 = (ACTIVE_MIN * 2 * Bsz * N_HEADS * Psz * D_HEAD) / 4;
    zero_prefix_kernel<<<(n4 + 255) / 256, 256, 0, stream>>>((float4*)out, n4);
  }
  {
    const int waves = L_ACT * 1024;                  // 12288 -> 1536 blocks
    gemm1_relu_kernel<<<waves / 8, 256, 0, stream>>>(Sh, W1t, b1, H);
  }
  {
    const int blocks = L_ACT * 256;                  // 3072 blocks, 256 thr
    gemm2_kv_kernel<<<blocks, 256, 0, stream>>>(H, W2t, b2, positions, out);
  }
}
